// HWRNNEncoder_70093866271017
// MI455X (gfx1250) — compile-verified, matrix operand order checked by reading, not run on hardware
//
#include <hip/hip_runtime.h>
#include <hip/hip_bf16.h>

#define B_ 32
#define S_ 384
#define D_ 512
#define H_ 512
#define M_ (S_*B_)      // 12288 rows (time-major)
#define NCAT 3072       // 4H (gates) + H (Wg x-part) + H (Wt)
#define NREC 2560       // 4H (Whh) + H (Wg h-part)
#define HS 520          // LDS row stride for h (bf16), pads banks

typedef __attribute__((ext_vector_type(16))) __bf16 v16bf;
typedef __attribute__((ext_vector_type(8)))  float  v8f;
typedef __attribute__((ext_vector_type(4)))  unsigned uv4;

union AF { uv4 u[2]; v16bf v; };

__device__ __forceinline__ unsigned short f2bf(float f) {
  unsigned u = __builtin_bit_cast(unsigned, f);
  u += 0x7FFFu + ((u >> 16) & 1u);
  return (unsigned short)(u >> 16);
}
__device__ __forceinline__ float sigf(float x) { return 1.0f / (1.0f + __expf(-x)); }

// ---------------- x: (B,S,D) f32 -> time-major [S*B][D] bf16 ----------------
__global__ void xcvt_kernel(const float* __restrict__ x, unsigned short* __restrict__ X0) {
  size_t idx = (size_t)blockIdx.x * blockDim.x + threadIdx.x;
  if (idx >= (size_t)M_ * D_) return;
  int c = (int)(idx % D_);
  int row = (int)(idx / D_);
  int t = row / B_, b = row % B_;
  X0[idx] = f2bf(x[((size_t)b * S_ + t) * D_ + c]);
}

// -------- pack [Wih;Wg_x;Wt] (NCAT x K) into WMMA B fragments (bf16) --------
// fragment(nt,kt): 32 lanes x 16 bf16; lane l: n = nt*16+(l&15), hi=l>>4;
// slot s: v=s>>1, group=v>>2, r=((v&3)<<1)|(s&1), k = kt*32+group*16+hi*8+r
__global__ void pack_wcat_kernel(const float* __restrict__ Wih,
                                 const float* __restrict__ Wg,
                                 const float* __restrict__ Wt,
                                 int K, unsigned short* __restrict__ dst, int total) {
  int idx = blockIdx.x * blockDim.x + threadIdx.x;
  if (idx >= total) return;
  int numKt = K >> 5;
  int frag = idx >> 9, within = idx & 511;
  int l = within >> 4, s = within & 15;
  int nt = frag / numKt, kt = frag % numKt;
  int v = s >> 1, half = s & 1;
  int group = v >> 2, r = ((v & 3) << 1) | half;
  int hi = l >> 4;
  int k = kt * 32 + group * 16 + hi * 8 + r;
  int n = nt * 16 + (l & 15);
  float val;
  if (n < 4 * H_)      val = Wih[(size_t)n * K + k];
  else if (n < 5 * H_) val = Wg[(size_t)(n - 4 * H_) * (K + H_) + k];  // Wg row len = din+H
  else                 val = Wt[(size_t)(n - 5 * H_) * K + k];
  dst[idx] = f2bf(val);
}

// -------- pack [Whh;Wg_h] (NREC x H) into WMMA B fragments (bf16) ----------
__global__ void pack_wr_kernel(const float* __restrict__ Whh,
                               const float* __restrict__ Wg,
                               int din, unsigned short* __restrict__ dst, int total) {
  int idx = blockIdx.x * blockDim.x + threadIdx.x;
  if (idx >= total) return;
  const int numKt = H_ >> 5;  // 16
  int frag = idx >> 9, within = idx & 511;
  int l = within >> 4, s = within & 15;
  int nt = frag / numKt, kt = frag % numKt;
  int v = s >> 1, half = s & 1;
  int group = v >> 2, r = ((v & 3) << 1) | half;
  int hi = l >> 4;
  int k = kt * 32 + group * 16 + hi * 8 + r;
  int n = nt * 16 + (l & 15);
  float val;
  if (n < 4 * H_) val = Whh[(size_t)n * H_ + k];
  else            val = Wg[(size_t)(n - 4 * H_) * (din + H_) + din + k];
  dst[idx] = f2bf(val);
}

// ---------------- fused bias: [bih+bhh ; bg ; 0] ----------------
__global__ void biascat_kernel(const float* __restrict__ bih, const float* __restrict__ bhh,
                               const float* __restrict__ bg, float* __restrict__ dst) {
  int j = blockIdx.x * blockDim.x + threadIdx.x;
  if (j >= NCAT) return;
  float v = 0.0f;
  if (j < 4 * H_)      v = bih[j] + bhh[j];
  else if (j < 5 * H_) v = bg[j - 4 * H_];
  dst[j] = v;
}

// ---------------- time-parallel GEMM: P = X @ Wcat^T + bias ----------------
// grid (M/128, NCAT/64, 2 dirs); block 256 (8 waves); wave: 1 m-tile x 4 n-tiles
__global__ void __launch_bounds__(256) gemm_pre_kernel(
    const unsigned short* __restrict__ A, int K,
    const unsigned short* __restrict__ Bp_fw, const unsigned short* __restrict__ Bp_bw,
    const float* __restrict__ bias_fw, const float* __restrict__ bias_bw,
    float* __restrict__ C_fw, float* __restrict__ C_bw) {
  const int dir = blockIdx.z;
  const unsigned short* Bp = dir ? Bp_bw : Bp_fw;
  const float* bias = dir ? bias_bw : bias_fw;
  float* C = dir ? C_bw : C_fw;

  const int lane = threadIdx.x & 31;
  const int wave = threadIdx.x >> 5;
  const int mt = blockIdx.x * 8 + wave;
  const int nt0 = blockIdx.y * 4;
  const int numKt = K >> 5;
  const int hi = lane >> 4, lo = lane & 15;

  v8f acc[4];
#pragma unroll
  for (int j = 0; j < 4; ++j) acc[j] = (v8f){};

  const int arow = mt * 16 + lo;
  for (int kt = 0; kt < numKt; ++kt) {
    AF a;
    const uv4* pa = reinterpret_cast<const uv4*>(A + (size_t)arow * K + kt * 32 + hi * 8);
    a.u[0] = pa[0];
    a.u[1] = pa[2];  // +16 elements
#pragma unroll
    for (int j = 0; j < 4; ++j) {
      AF bf;
      const uv4* pb = reinterpret_cast<const uv4*>(
          Bp + (((size_t)(nt0 + j) * numKt + kt) * 32 + lane) * 16);
      bf.u[0] = pb[0];
      bf.u[1] = pb[1];
      acc[j] = __builtin_amdgcn_wmma_f32_16x16x32_bf16(false, a.v, false, bf.v,
                                                       (short)0, acc[j], false, false);
    }
  }
#pragma unroll
  for (int j = 0; j < 4; ++j) {
    int col = (nt0 + j) * 16 + lo;
    float bv = bias[col];
#pragma unroll
    for (int r = 0; r < 8; ++r) {
      int row = mt * 16 + hi * 8 + r;
      C[(size_t)row * NCAT + col] = acc[j][r] + bv;
    }
  }
}

// ---------------- sequential recurrent kernel ----------------
// grid 2 blocks (dir), block 1024 (32 waves). Wave w owns columns [16w,16w+16)
// of i,f,g,o,gate: B tiles {w, 32+w, 64+w, 96+w, 128+w} x 2 row-tiles.
__global__ void __launch_bounds__(1024) seq_kernel(
    const float* __restrict__ P_fw, const float* __restrict__ P_bw,
    const unsigned short* __restrict__ Wr_fw, const unsigned short* __restrict__ Wr_bw,
    const float* __restrict__ mask,
    float* __restrict__ outf,           // [S][B][1024] f32 or null
    unsigned short* __restrict__ outbf, // [S][B][1024] bf16 or null
    float* __restrict__ hnb, float* __restrict__ cnb) {
  const int dir = blockIdx.x;
  const float* P = dir ? P_bw : P_fw;
  const unsigned short* Wr = dir ? Wr_bw : Wr_fw;

  __shared__ __align__(16) unsigned short hsh[32 * HS];

  const int tid = threadIdx.x;
  const int lane = tid & 31, wave = tid >> 5;
  const int hi = lane >> 4, lo = lane & 15;
  const int cw = wave * 16 + lo;  // owned column in [0,512)

  for (int i = tid; i < 32 * HS; i += 1024) hsh[i] = 0;
  __syncthreads();

  v8f c0 = (v8f){}, c1 = (v8f){};

  for (int s = 0; s < S_; ++s) {
    const int t = dir ? (S_ - 1 - s) : s;

    v8f acc[5][2];
#pragma unroll
    for (int j = 0; j < 5; ++j) { acc[j][0] = (v8f){}; acc[j][1] = (v8f){}; }

#pragma unroll 4
    for (int kt = 0; kt < 16; ++kt) {
      AF a0, a1;
      const uv4* p0 = reinterpret_cast<const uv4*>(&hsh[(size_t)lo * HS + kt * 32 + hi * 8]);
      a0.u[0] = p0[0]; a0.u[1] = p0[2];
      const uv4* p1 = reinterpret_cast<const uv4*>(&hsh[(size_t)(16 + lo) * HS + kt * 32 + hi * 8]);
      a1.u[0] = p1[0]; a1.u[1] = p1[2];
#pragma unroll
      for (int j = 0; j < 5; ++j) {
        const int nt = j * 32 + wave;
        AF bf;
        const uv4* pb = reinterpret_cast<const uv4*>(Wr + (((size_t)nt * 16 + kt) * 32 + lane) * 16);
        bf.u[0] = pb[0];
        bf.u[1] = pb[1];
        acc[j][0] = __builtin_amdgcn_wmma_f32_16x16x32_bf16(false, a0.v, false, bf.v,
                                                            (short)0, acc[j][0], false, false);
        acc[j][1] = __builtin_amdgcn_wmma_f32_16x16x32_bf16(false, a1.v, false, bf.v,
                                                            (short)0, acc[j][1], false, false);
      }
    }

    float htv[2][8], cnv[2][8];
#pragma unroll
    for (int m = 0; m < 2; ++m) {
#pragma unroll
      for (int r = 0; r < 8; ++r) {
        const int b = m * 16 + hi * 8 + r;
        const float* prow = P + ((size_t)t * B_ + b) * NCAT;
        float gi = sigf(acc[0][m][r] + prow[cw]);
        float gf = sigf(acc[1][m][r] + prow[512 + cw]);
        float gg = tanhf(acc[2][m][r] + prow[1024 + cw]);
        float go = sigf(acc[3][m][r] + prow[1536 + cw]);
        float cp = m ? c1[r] : c0[r];
        float cn = gf * cp + gi * gg;
        float hn = go * tanhf(cn);
        float mt_ = mask[(size_t)b * S_ + t];
        hn *= mt_;
        cn *= mt_;
        float gate = sigf(acc[4][m][r] + prow[2048 + cw]);
        float tp = prow[2560 + cw];
        float ht = gate * hn + (1.0f - gate) * tp;
        if (m) c1[r] = cn; else c0[r] = cn;
        htv[m][r] = ht;
        cnv[m][r] = cn;
      }
    }

    __syncthreads();  // everyone done reading old h
#pragma unroll
    for (int m = 0; m < 2; ++m) {
#pragma unroll
      for (int r = 0; r < 8; ++r) {
        const int b = m * 16 + hi * 8 + r;
        float ht = htv[m][r];
        hsh[b * HS + cw] = f2bf(ht);
        size_t oidx = ((size_t)t * B_ + b) * 1024 + dir * 512 + cw;
        if (outf) outf[oidx] = ht;
        if (outbf) outbf[oidx] = f2bf(ht);
      }
    }
    if (s == S_ - 1) {
#pragma unroll
      for (int m = 0; m < 2; ++m)
#pragma unroll
        for (int r = 0; r < 8; ++r) {
          const int b = m * 16 + hi * 8 + r;
          hnb[(size_t)b * 1024 + dir * 512 + cw] = htv[m][r];
          cnb[(size_t)b * 1024 + dir * 512 + cw] = cnv[m][r];
        }
    }
    __syncthreads();
  }
}

// ---------------- [S][B][1024] -> (B,S,1024) ----------------
__global__ void out_tr_kernel(const float* __restrict__ src, float* __restrict__ dst) {
  size_t idx = (size_t)blockIdx.x * blockDim.x + threadIdx.x;
  if (idx >= (size_t)M_ * 1024) return;
  int c = (int)(idx & 1023);
  size_t row = idx >> 10;
  int t = (int)(row / B_), b = (int)(row % B_);
  dst[((size_t)b * S_ + t) * 1024 + c] = src[idx];
}

extern "C" void kernel_launch(void* const* d_in, const int* in_sizes, int n_in,
                              void* d_out, int out_size, void* d_ws, size_t ws_size,
                              hipStream_t stream) {
  (void)in_sizes; (void)n_in; (void)out_size; (void)ws_size;
  const float* x = (const float*)d_in[0];
  const float* mask = (const float*)d_in[1];
  // per (layer, dir): Wih, Whh, bih, bhh, Wg, bg, Wt
  auto W = [&](int l, int d, int i) -> const float* { return (const float*)d_in[2 + l * 14 + d * 7 + i]; };

  char* ws = (char*)d_ws;
  size_t off = 0;
  auto alloc = [&](size_t bytes) -> char* {
    char* p = ws + off;
    off += (bytes + 255) & ~(size_t)255;
    return p;
  };

  unsigned short* X0 = (unsigned short*)alloc((size_t)M_ * 512 * 2);
  unsigned short* X1 = (unsigned short*)alloc((size_t)M_ * 1024 * 2);
  unsigned short* Wcat[2][2];
  unsigned short* Wrp[2][2];
  float* bcat[2][2];
  for (int l = 0; l < 2; ++l)
    for (int d = 0; d < 2; ++d) {
      int din = l ? 1024 : 512;
      Wcat[l][d] = (unsigned short*)alloc((size_t)NCAT * din * 2);
      Wrp[l][d]  = (unsigned short*)alloc((size_t)NREC * H_ * 2);
      bcat[l][d] = (float*)alloc((size_t)NCAT * 4);
    }
  float* P0 = (float*)alloc((size_t)M_ * NCAT * 4);
  float* P1 = (float*)alloc((size_t)M_ * NCAT * 4);
  float* out1f = (float*)alloc((size_t)M_ * 1024 * 4);

  const size_t OUTE = (size_t)B_ * S_ * 1024;  // 12,582,912
  float* outp = (float*)d_out;
  float* hn_base = outp + OUTE;                 // (2, B, 1024)
  float* cn_base = hn_base + (size_t)2 * B_ * 1024;

  // 1) x -> bf16 time-major
  {
    size_t total = (size_t)M_ * 512;
    xcvt_kernel<<<dim3((unsigned)((total + 255) / 256)), 256, 0, stream>>>(x, X0);
  }
  // 2) weight repack + fused biases
  for (int l = 0; l < 2; ++l)
    for (int d = 0; d < 2; ++d) {
      int din = l ? 1024 : 512;
      int totc = NCAT * din;
      pack_wcat_kernel<<<dim3((totc + 255) / 256), 256, 0, stream>>>(
          W(l, d, 0), W(l, d, 4), W(l, d, 6), din, Wcat[l][d], totc);
      int totr = NREC * H_;
      pack_wr_kernel<<<dim3((totr + 255) / 256), 256, 0, stream>>>(
          W(l, d, 1), W(l, d, 4), din, Wrp[l][d], totr);
      biascat_kernel<<<dim3((NCAT + 255) / 256), 256, 0, stream>>>(
          W(l, d, 2), W(l, d, 3), W(l, d, 5), bcat[l][d]);
    }
  // 3) layer 0: batched GEMM (both dirs), then sequential (both dirs concurrent)
  gemm_pre_kernel<<<dim3(M_ / 128, NCAT / 64, 2), 256, 0, stream>>>(
      X0, 512, Wcat[0][0], Wcat[0][1], bcat[0][0], bcat[0][1], P0, P1);
  seq_kernel<<<dim3(2), 1024, 0, stream>>>(
      P0, P1, Wrp[0][0], Wrp[0][1], mask,
      (float*)nullptr, X1, hn_base + 0, cn_base + 0);
  // 4) layer 1
  gemm_pre_kernel<<<dim3(M_ / 128, NCAT / 64, 2), 256, 0, stream>>>(
      X1, 1024, Wcat[1][0], Wcat[1][1], bcat[1][0], bcat[1][1], P0, P1);
  seq_kernel<<<dim3(2), 1024, 0, stream>>>(
      P0, P1, Wrp[1][0], Wrp[1][1], mask,
      out1f, (unsigned short*)nullptr,
      hn_base + (size_t)B_ * 1024, cn_base + (size_t)B_ * 1024);
  // 5) final transpose to (B,S,2H)
  {
    size_t total = (size_t)M_ * 1024;
    out_tr_kernel<<<dim3((unsigned)((total + 255) / 256)), 256, 0, stream>>>(out1f, outp);
  }
}
